// ProposalTarget_32847909879810
// MI455X (gfx1250) — matrix operand order repaired and checked by reading, not compiled, not run on hardware
//
#include <hip/hip_runtime.h>
#include <hip/hip_bf16.h>

// ---------------- constants (match reference) ----------------
#define NPROP 80000
#define KGT   256
#define NCLS  81
#define MTOT  (NPROP + KGT)      // 80256 (divisible by 32)
#define BSZ   256
#define FGROIS 128
#define FG_THRESH 0.7f
#define BG_HI 0.5f
#define BG_LO 0.1f

typedef __attribute__((ext_vector_type(16))) _Float16 v16h;
typedef __attribute__((ext_vector_type(8)))  float    v8f;

// ---------------- K1: IoU + row argmax over K=256 gt boxes ----------------
__global__ void k_iou_argmax(const float* __restrict__ proposals,
                             const float* __restrict__ gt_boxes,
                             float* __restrict__ max_ov,
                             int* __restrict__ gt_assign) {
    __shared__ float4 sgt[KGT];
    for (int k = threadIdx.x; k < KGT; k += blockDim.x)
        sgt[k] = reinterpret_cast<const float4*>(gt_boxes)[k];
    __syncthreads();

    int i = blockIdx.x * blockDim.x + threadIdx.x;
    if (i >= MTOT) return;

    // hint next block's proposals into cache (global_prefetch_b8)
    if (i + 256 < NPROP) __builtin_prefetch(&proposals[(size_t)(i + 256) * 4], 0, 1);

    float4 a = (i < NPROP) ? reinterpret_cast<const float4*>(proposals)[i]
                           : reinterpret_cast<const float4*>(gt_boxes)[i - NPROP];
    float area_a = (a.z - a.x + 1.0f) * (a.w - a.y + 1.0f);

    float best = -1.0f;  // IoU >= 0 so first k wins ties (jnp.argmax semantics)
    int besti = 0;
    for (int k = 0; k < KGT; ++k) {
        float4 b = sgt[k];
        float area_b = (b.z - b.x + 1.0f) * (b.w - b.y + 1.0f);
        float x1 = fmaxf(a.x, b.x), y1 = fmaxf(a.y, b.y);
        float x2 = fminf(a.z, b.z), y2 = fminf(a.w, b.w);
        float iw = fmaxf(x2 - x1 + 1.0f, 0.0f);
        float ih = fmaxf(y2 - y1 + 1.0f, 0.0f);
        float inter = iw * ih;
        float iou = inter / (area_a + area_b - inter);
        if (iou > best) { best = iou; besti = k; }
    }
    max_ov[i] = best;
    gt_assign[i] = besti;
}

// ---------------- K2: wave32 ordered compaction (stable partition) ----------------
// fg_list: first 128 indices with max_ov>=0.7, in index order
// bg_list: first 256 indices with 0.1<=max_ov<0.5, in index order
// nonbg_list: first 256 indices NOT bg, in index order (fallback of bg_sorted)
// counts = {num_fg, num_bg, fg_take}
__global__ void k_select(const float* __restrict__ max_ov,
                         int* __restrict__ fg_list,
                         int* __restrict__ bg_list,
                         int* __restrict__ nonbg_list,
                         int* __restrict__ counts) {
    const int lane = threadIdx.x;                 // 32 threads, one wave
    const unsigned long long lm = (1ull << lane) - 1ull;
    int fg_cnt = 0, bg_cnt = 0, nb_cnt = 0;
    for (int base = 0; base < MTOT; base += 32) { // MTOT % 32 == 0
        int i = base + lane;
        float v = max_ov[i];
        bool fg = (v >= FG_THRESH);
        bool bg = (v < BG_HI) && (v >= BG_LO);
        bool nb = !bg;
        unsigned long long mfg = __ballot(fg);
        unsigned long long mbg = __ballot(bg);
        unsigned long long mnb = __ballot(nb);
        if (fg) { int idx = fg_cnt + __popcll(mfg & lm); if (idx < FGROIS) fg_list[idx] = i; }
        if (bg) { int idx = bg_cnt + __popcll(mbg & lm); if (idx < BSZ)    bg_list[idx] = i; }
        if (nb) { int idx = nb_cnt + __popcll(mnb & lm); if (idx < BSZ)    nonbg_list[idx] = i; }
        fg_cnt += (int)__popcll(mfg);
        bg_cnt += (int)__popcll(mbg);
        nb_cnt += (int)__popcll(mnb);
    }
    if (lane == 0) {
        counts[0] = fg_cnt;
        counts[1] = bg_cnt;
        counts[2] = (fg_cnt < FGROIS) ? fg_cnt : FGROIS;  // fg_take
    }
}

// ---------------- K3: build keep, rois, regression targets, class ids ----------------
__global__ void k_gather(const float* __restrict__ proposals,
                         const float* __restrict__ gt_boxes,
                         const float* __restrict__ gt_labels,
                         const int* __restrict__ gt_assign,
                         const int* __restrict__ fg_list,
                         const int* __restrict__ bg_list,
                         const int* __restrict__ nonbg_list,
                         const int* __restrict__ counts,
                         float* __restrict__ out_rois,
                         int* __restrict__ assign_keep,
                         int* __restrict__ cls_keep,
                         float* __restrict__ targets) {
    int p = threadIdx.x;            // 256 threads
    int fg_take = counts[2];
    int num_bg  = counts[1];
    bool is_fg = (p < fg_take);
    int keep;
    if (is_fg) {
        keep = fg_list[p];
    } else {
        int j = p - fg_take;
        keep = (j < num_bg) ? bg_list[j] : nonbg_list[j - num_bg];
    }

    float4 ex = (keep < NPROP) ? reinterpret_cast<const float4*>(proposals)[keep]
                               : reinterpret_cast<const float4*>(gt_boxes)[keep - NPROP];
    int ga = gt_assign[keep];
    float4 gt = reinterpret_cast<const float4*>(gt_boxes)[ga];

    out_rois[p * 4 + 0] = ex.x;
    out_rois[p * 4 + 1] = ex.y;
    out_rois[p * 4 + 2] = ex.z;
    out_rois[p * 4 + 3] = ex.w;
    assign_keep[p] = ga;

    // bbox transform (+1 convention)
    float ew = ex.z - ex.x + 1.0f, eh = ex.w - ex.y + 1.0f;
    float ecx = ex.x + 0.5f * ew,  ecy = ex.y + 0.5f * eh;
    float gw = gt.z - gt.x + 1.0f, gh = gt.w - gt.y + 1.0f;
    float gcx = gt.x + 0.5f * gw,  gcy = gt.y + 0.5f * gh;
    targets[p * 4 + 0] = (gcx - ecx) / ew;
    targets[p * 4 + 1] = (gcy - ecy) / eh;
    targets[p * 4 + 2] = __logf(gw / ew);
    targets[p * 4 + 3] = __logf(gh / eh);

    // class = argmax over gt_labels row (first max); bg slots -> class 0
    int cls = 0; float bv = -1.0f;
    for (int c = 0; c < NCLS; ++c) {
        float v = gt_labels[ga * NCLS + c];
        if (v > bv) { bv = v; cls = c; }
    }
    cls_keep[p] = is_fg ? cls : 0;
}

// ---------------- K4: labels via WMMA  D(256x96) = Sel(256x256) x gt_labels(256x81 pad 96) ----
// Sel[b,k] = (assign_keep[b]==k); {0,1} operands -> exact in f16.
// 1 block, 512 threads = 16 waves; wave w owns rows 16w..16w+15.
// B is staged once in LDS as transposed f16 (96 cols x 256 K), rows 32B-aligned
// and bank-skewed, so each B fragment is one aligned v16h (2x ds_load_b128).
#define BSTRIDE 272   // f16 per LDS row: 544 B -> 32B-aligned rows, skewed banks

__global__ void k_labels_wmma(const float* __restrict__ gt_labels,
                              const int* __restrict__ assign_keep,
                              const int* __restrict__ counts,
                              float* __restrict__ out_labels) {
    __shared__ __align__(32) _Float16 sB[96 * BSTRIDE];   // ~51 KB of 320 KB/WGP

    // stage gt_labels^T as f16; zero-pad cols 81..95 (coalesced global reads)
    for (int idx = threadIdx.x; idx < 96 * KGT; idx += 512) {
        int n = idx % 96;          // padded class column
        int k = idx / 96;          // gt index (K dimension)
        float v = (n < NCLS) ? gt_labels[k * NCLS + n] : 0.0f;
        sB[n * BSTRIDE + k] = (_Float16)v;
    }
    __syncthreads();

    const int lane = threadIdx.x & 31;
    const int wave = threadIdx.x >> 5;      // row tile 0..15
    const int half = lane >> 4;             // 0/1
    const int l16  = lane & 15;
    const int fg_take = counts[2];
    const int arow_assign = assign_keep[wave * 16 + l16];  // A row held by this lane

    v8f zero = {};
    v8f acc0 = zero, acc1 = zero, acc2 = zero, acc3 = zero, acc4 = zero, acc5 = zero;

#pragma unroll
    for (int kk = 0; kk < 8; ++kk) {        // K = 256 in 8 steps of 32
        // A 16x32 f16 layout: K = 32kk + 16*half + e
        v16h a;
#pragma unroll
        for (int e = 0; e < 16; ++e) {
            int ka = 32 * kk + 16 * half + e;
            a[e] = (arow_assign == ka) ? (_Float16)1.0f : (_Float16)0.0f;
        }
        // B 32x16 f16 layout: lane holds col n, K = 32kk + 16*half + e
        const int kb = kk * 32 + half * 16;
        const v16h b0 = *(const v16h*)&sB[(0 * 16 + l16) * BSTRIDE + kb];
        const v16h b1 = *(const v16h*)&sB[(1 * 16 + l16) * BSTRIDE + kb];
        const v16h b2 = *(const v16h*)&sB[(2 * 16 + l16) * BSTRIDE + kb];
        const v16h b3 = *(const v16h*)&sB[(3 * 16 + l16) * BSTRIDE + kb];
        const v16h b4 = *(const v16h*)&sB[(4 * 16 + l16) * BSTRIDE + kb];
        const v16h b5 = *(const v16h*)&sB[(5 * 16 + l16) * BSTRIDE + kb];
        acc0 = __builtin_amdgcn_wmma_f32_16x16x32_f16(false, a, false, b0, (short)0, acc0, false, false);
        acc1 = __builtin_amdgcn_wmma_f32_16x16x32_f16(false, a, false, b1, (short)0, acc1, false, false);
        acc2 = __builtin_amdgcn_wmma_f32_16x16x32_f16(false, a, false, b2, (short)0, acc2, false, false);
        acc3 = __builtin_amdgcn_wmma_f32_16x16x32_f16(false, a, false, b3, (short)0, acc3, false, false);
        acc4 = __builtin_amdgcn_wmma_f32_16x16x32_f16(false, a, false, b4, (short)0, acc4, false, false);
        acc5 = __builtin_amdgcn_wmma_f32_16x16x32_f16(false, a, false, b5, (short)0, acc5, false, false);
    }

    // D layout: acc[i] -> row = 16*wave + i + 8*half, col = 16*t + l16
    v8f accs[6] = {acc0, acc1, acc2, acc3, acc4, acc5};
#pragma unroll
    for (int t = 0; t < 6; ++t) {
        int n = t * 16 + l16;
        if (n < NCLS) {
#pragma unroll
            for (int i = 0; i < 8; ++i) {
                int m = wave * 16 + i + 8 * half;
                float v = (m < fg_take) ? accs[t][i] : (n == 0 ? 1.0f : 0.0f);
                out_labels[m * NCLS + n] = v;
            }
        }
    }
}

// ---------------- K5: expand targets to (256 x 4*81), zeros elsewhere ----------------
__global__ void k_bbox(const int* __restrict__ cls_keep,
                       const int* __restrict__ counts,
                       const float* __restrict__ targets,
                       float* __restrict__ out_tg) {
    int idx = blockIdx.x * blockDim.x + threadIdx.x;
    if (idx >= BSZ * 4 * NCLS) return;
    int b = idx / (4 * NCLS);
    int j = idx % (4 * NCLS);
    int c = j >> 2, comp = j & 3;
    float v = 0.0f;
    if (b < counts[2]) {
        int cl = cls_keep[b];
        if (cl > 0 && c == cl) v = targets[b * 4 + comp];
    }
    out_tg[idx] = v;
}

// ---------------- launcher ----------------
extern "C" void kernel_launch(void* const* d_in, const int* in_sizes, int n_in,
                              void* d_out, int out_size, void* d_ws, size_t ws_size,
                              hipStream_t stream) {
    const float* proposals = (const float*)d_in[0];   // (1, 80000, 4)
    const float* gt_labels = (const float*)d_in[1];   // (1, 256, 81)
    const float* gt_boxes  = (const float*)d_in[2];   // (1, 256, 4)

    char* ws = (char*)d_ws;
    size_t off = 0;
    float* max_ov     = (float*)(ws + off); off += (size_t)MTOT * sizeof(float);
    int*   gt_assign  = (int*)  (ws + off); off += (size_t)MTOT * sizeof(int);
    int*   fg_list    = (int*)  (ws + off); off += FGROIS * sizeof(int);
    int*   bg_list    = (int*)  (ws + off); off += BSZ * sizeof(int);
    int*   nonbg_list = (int*)  (ws + off); off += BSZ * sizeof(int);
    int*   counts     = (int*)  (ws + off); off += 8 * sizeof(int);
    int*   assign_keep= (int*)  (ws + off); off += BSZ * sizeof(int);
    int*   cls_keep   = (int*)  (ws + off); off += BSZ * sizeof(int);
    float* targets    = (float*)(ws + off); off += BSZ * 4 * sizeof(float);

    float* out_rois   = (float*)d_out;                 // 256*4
    float* out_labels = out_rois + BSZ * 4;            // 256*81
    float* out_tg     = out_labels + BSZ * NCLS;       // 256*324

    k_iou_argmax<<<(MTOT + 255) / 256, 256, 0, stream>>>(proposals, gt_boxes, max_ov, gt_assign);
    k_select<<<1, 32, 0, stream>>>(max_ov, fg_list, bg_list, nonbg_list, counts);
    k_gather<<<1, 256, 0, stream>>>(proposals, gt_boxes, gt_labels, gt_assign,
                                    fg_list, bg_list, nonbg_list, counts,
                                    out_rois, assign_keep, cls_keep, targets);
    k_labels_wmma<<<1, 512, 0, stream>>>(gt_labels, assign_keep, counts, out_labels);
    k_bbox<<<(BSZ * 4 * NCLS + 255) / 256, 256, 0, stream>>>(cls_keep, counts, targets, out_tg);
}
